// RelationTransformerDecoderLayer_1769526526261
// MI455X (gfx1250) — compile-verified, matrix operand order checked by reading, not run on hardware
//
#include <hip/hip_runtime.h>
#include <hip/hip_bf16.h>

// ---------------- problem constants ----------------
#define BB   8
#define NQ   900
#define NQP  928          // NQ padded to a multiple of 32 for the WMMA K loop
#define DD   256
#define HH   8
#define DH   32
#define NH   4
#define RH   4
#define LL   4
#define PP   4
#define LV   21760
#define DFF  1024
#define BNQ  (BB*NQ)      // 7200
#define BLV  (BB*LV)      // 174080
#define SCALE_QK 0.1767766952966369f   // 32^-0.5

typedef __attribute__((ext_vector_type(16))) __bf16 v16bf;
typedef __attribute__((ext_vector_type(8)))  __bf16 v8bf;
typedef __attribute__((ext_vector_type(8)))  float  v8f;
typedef __attribute__((ext_vector_type(4)))  float  v4f;

// ---------------- WMMA fragment loaders (bf16 16x16x32) ----------------
// A fragment from fp32 row-major (lda floats). Lane l<16: row l, K {k0..k0+7} in
// frag[0..7], {k0+16..k0+23} in frag[8..15]; lanes 16..31 hold K {+8..15,+24..31}.
// Out-of-range rows are CLAMPED (caller masks at store / relies on zeroed operand).
__device__ __forceinline__ v16bf load_A_f32(const float* __restrict__ A, int lda,
                                            int row0, int k0, int Mlim) {
  const int lane = threadIdx.x & 31;
  int r = row0 + (lane & 15);
  r = (r < Mlim) ? r : (Mlim - 1);
  const int hi = lane >> 4;
  const int ka = k0 + (hi ? 8 : 0);
  const int kb = k0 + (hi ? 24 : 16);
  const float* p = A + (long)r * lda;
  const v4f a0 = *(const v4f*)(p + ka);
  const v4f a1 = *(const v4f*)(p + ka + 4);
  const v4f b0 = *(const v4f*)(p + kb);
  const v4f b1 = *(const v4f*)(p + kb + 4);
  v16bf f;
#pragma unroll
  for (int i = 0; i < 4; ++i) {
    f[i]      = (__bf16)a0[i];
    f[4 + i]  = (__bf16)a1[i];
    f[8 + i]  = (__bf16)b0[i];
    f[12 + i] = (__bf16)b1[i];
  }
  return f;
}

// B fragment from bf16 "transposed" N x K storage (ldk bf16 elements per row).
// Element (k, n) = WT[n*ldk + k]; two contiguous 16B loads.
__device__ __forceinline__ v16bf load_B_t(const __bf16* __restrict__ WT, int ldk,
                                          int k0, int col0) {
  const int lane = threadIdx.x & 31;
  const int n = col0 + (lane & 15);
  const int hi = lane >> 4;
  const int ka = k0 + (hi ? 8 : 0);
  const int kb = k0 + (hi ? 24 : 16);
  const __bf16* p = WT + (long)n * ldk;
  const v8bf lo = *(const v8bf*)(p + ka);
  const v8bf hi8 = *(const v8bf*)(p + kb);
  v16bf f;
#pragma unroll
  for (int i = 0; i < 8; ++i) { f[i] = lo[i]; f[8 + i] = hi8[i]; }
  return f;
}

// ---------------- weight transpose + bf16 convert: WT[n*K+k] = W[k*N+n] ----------------
__global__ void transpose_w_kernel(const float* __restrict__ W, __bf16* __restrict__ WT,
                                   int K, int N) {
  const int k = blockIdx.x * 32 + threadIdx.x;
  const int n = blockIdx.y * 8 + threadIdx.y;
  if (k < K && n < N) WT[(long)n * K + k] = (__bf16)W[(long)k * N + n];
}

// ---------------- pack V into per-(b,h) bf16 (DH x NQP) transposed tiles ----------------
__global__ void pack_v_kernel(const float* __restrict__ V, __bf16* __restrict__ VT) {
  const int k  = blockIdx.x * 32 + threadIdx.x;   // key index, 0..NQP-1
  const int bh = blockIdx.y;
  const int b = bh / HH, h = bh % HH;
#pragma unroll
  for (int n0 = 0; n0 < DH; n0 += 8) {
    const int n = n0 + threadIdx.y;
    const float vv = (k < NQ) ? V[((long)b * NQ + k) * DD + h * DH + n] : 0.0f;
    VT[((long)bh * DH + n) * NQP + k] = (__bf16)vv;
  }
}

// ---------------- WMMA GEMM: C[M,N] = A[M,K] @ W[K,N] + bias ----------------
// One wave computes a 16x64 tile (4 accumulators, A fragment reused 4x).
// Requires M % 16 == 0 and N % 64 == 0 (true for every projection here).
template <int KK>
__global__ void gemm_bias_kernel(const float* __restrict__ A, const __bf16* __restrict__ WT,
                                 const float* __restrict__ bias, float* __restrict__ C,
                                 int M, int N, int relu) {
  const int row0 = blockIdx.x * 16;
  const int col0 = blockIdx.y * 64;
  v8f acc[4] = {};
#pragma unroll 4
  for (int k0 = 0; k0 < KK; k0 += 32) {
    const v16bf a = load_A_f32(A, KK, row0, k0, 0x7fffffff);
#pragma unroll
    for (int c = 0; c < 4; ++c) {
      const v16bf b = load_B_t(WT, KK, k0, col0 + c * 16);
      acc[c] = __builtin_amdgcn_wmma_f32_16x16x32_bf16(false, a, false, b, (short)0, acc[c], false, false);
    }
  }
  const int lane = threadIdx.x & 31;
  const int mofs = (lane >> 4) ? 8 : 0;
#pragma unroll
  for (int c = 0; c < 4; ++c) {
    const int n = col0 + c * 16 + (lane & 15);
    const float bv = bias ? bias[n] : 0.0f;
#pragma unroll
    for (int i = 0; i < 8; ++i) {
      float v = acc[c][i] + bv;
      if (relu) v = fmaxf(v, 0.0f);
      C[(long)(row0 + mofs + i) * N + n] = v;
    }
  }
}

// ---------------- attention scores: S_bh = (Q_bh @ K_bh^T) * scale ----------------
// One wave computes 16x32 (2 accumulators). Both operands are fp32 with contiguous
// K (lda = DD); the A/B lane layouts are symmetric so load_A_f32 serves both (B:
// "row" = key column, clamped to NQ-1 -- garbage pad columns are overwritten by the
// softmax kernel; garbage pad rows are never stored).
__global__ void attn_scores_kernel(const float* __restrict__ Q, const float* __restrict__ Km,
                                   float* __restrict__ S) {
  const int bh = blockIdx.z;
  const int b = bh / HH, h = bh % HH;
  const int row0 = blockIdx.x * 16;
  const int col0 = blockIdx.y * 32;
  const float* Qb = Q + (long)b * NQ * DD + h * DH;
  const float* Kb = Km + (long)b * NQ * DD + h * DH;
  const v16bf a = load_A_f32(Qb, DD, row0, 0, NQ);
  v8f acc[2] = {};
#pragma unroll
  for (int c = 0; c < 2; ++c) {
    const v16bf bfr = load_A_f32(Kb, DD, col0 + c * 16, 0, NQ);
    acc[c] = __builtin_amdgcn_wmma_f32_16x16x32_bf16(false, a, false, bfr, (short)0, acc[c], false, false);
  }
  float* Sb = S + (long)bh * NQ * NQP;
  const int lane = threadIdx.x & 31;
  const int mofs = (lane >> 4) ? 8 : 0;
  if (row0 + 16 <= NQ) {            // wave-uniform fast path (56 of 57 tiles)
#pragma unroll
    for (int c = 0; c < 2; ++c) {
      const int n = col0 + c * 16 + (lane & 15);
#pragma unroll
      for (int i = 0; i < 8; ++i)
        Sb[(long)(row0 + mofs + i) * NQP + n] = acc[c][i] * SCALE_QK;
    }
  } else {
#pragma unroll
    for (int c = 0; c < 2; ++c) {
      const int n = col0 + c * 16 + (lane & 15);
#pragma unroll
      for (int i = 0; i < 8; ++i) {
        const int m = row0 + mofs + i;
        if (m < NQ) Sb[(long)m * NQP + n] = acc[c][i] * SCALE_QK;
      }
    }
  }
}

// ---------------- softmax / relation normalization over each score row ----------------
__global__ void attn_softmax_kernel(float* __restrict__ S, const float* __restrict__ rel) {
  const int q  = blockIdx.x;
  const int bh = blockIdx.y;
  const int b = bh / HH, h = bh % HH;
  float* row = S + (long)bh * NQ * NQP + (long)q * NQP;
  __shared__ float red[256];
  const int t = threadIdx.x;
  float ev[4];
  float sum = 0.0f;
  if (h < NH) {
    float mx = -1e30f;
    for (int i = t; i < NQ; i += 256) mx = fmaxf(mx, row[i]);
    red[t] = mx; __syncthreads();
    for (int s2 = 128; s2 > 0; s2 >>= 1) { if (t < s2) red[t] = fmaxf(red[t], red[t + s2]); __syncthreads(); }
    mx = red[0]; __syncthreads();
#pragma unroll
    for (int j = 0; j < 4; ++j) {
      const int i = t + j * 256;
      ev[j] = (i < NQ) ? expf(row[i] - mx) : 0.0f;
      sum += ev[j];
    }
    red[t] = sum; __syncthreads();
    for (int s2 = 128; s2 > 0; s2 >>= 1) { if (t < s2) red[t] += red[t + s2]; __syncthreads(); }
    const float inv = 1.0f / red[0];
#pragma unroll
    for (int j = 0; j < 4; ++j) { const int i = t + j * 256; if (i < NQ) row[i] = ev[j] * inv; }
  } else {
    const float* rl = rel + (((long)b * RH + (h - NH)) * NQ + q) * (long)NQ;
#pragma unroll
    for (int j = 0; j < 4; ++j) {
      const int i = t + j * 256;
      ev[j] = (i < NQ) ? expf(row[i]) * rl[i] : 0.0f;
      sum += ev[j];
    }
    red[t] = sum; __syncthreads();
    for (int s2 = 128; s2 > 0; s2 >>= 1) { if (t < s2) red[t] += red[t + s2]; __syncthreads(); }
    const float inv = 1.0f / (red[0] + 1e-6f);
#pragma unroll
    for (int j = 0; j < 4; ++j) { const int i = t + j * 256; if (i < NQ) row[i] = ev[j] * inv; }
  }
  for (int i = NQ + t; i < NQP; i += 256) row[i] = 0.0f;   // zero the pad columns
}

// ---------------- O_bh = A_bh @ V_bh (bf16-packed V), stored interleaved (B,NQ,D) -------
__global__ void attn_v_kernel(const float* __restrict__ S, const __bf16* __restrict__ VT,
                              float* __restrict__ O) {
  const int bh = blockIdx.z;
  const int b = bh / HH, h = bh % HH;
  const int row0 = blockIdx.x * 16;
  const float* Sb = S + (long)bh * NQ * NQP;
  const __bf16* Vb = VT + (long)bh * DH * NQP;
  v8f acc[2] = {};
#pragma unroll 4
  for (int k0 = 0; k0 < NQP; k0 += 32) {
    const v16bf a = load_A_f32(Sb, NQP, row0, k0, NQ);  // clamped rows masked at store
#pragma unroll
    for (int c = 0; c < 2; ++c) {
      const v16bf bfr = load_B_t(Vb, NQP, k0, c * 16);  // pad K rows are zeros
      acc[c] = __builtin_amdgcn_wmma_f32_16x16x32_bf16(false, a, false, bfr, (short)0, acc[c], false, false);
    }
  }
  float* Ob = O + (long)b * NQ * DD + h * DH;
  const int lane = threadIdx.x & 31;
  const int mofs = (lane >> 4) ? 8 : 0;
  if (row0 + 16 <= NQ) {            // wave-uniform fast path
#pragma unroll
    for (int c = 0; c < 2; ++c) {
      const int n = c * 16 + (lane & 15);
#pragma unroll
      for (int i = 0; i < 8; ++i)
        Ob[(long)(row0 + mofs + i) * DD + n] = acc[c][i];
    }
  } else {
#pragma unroll
    for (int c = 0; c < 2; ++c) {
      const int n = c * 16 + (lane & 15);
#pragma unroll
      for (int i = 0; i < 8; ++i) {
        const int m = row0 + mofs + i;
        if (m < NQ) Ob[(long)m * DD + n] = acc[c][i];
      }
    }
  }
}

// ---------------- layernorm: y = LN(x (+ r)) * w + b, one wave per 256-wide row ----------
__global__ void ln_kernel(const float* __restrict__ x, const float* __restrict__ r,
                          const float* __restrict__ w, const float* __restrict__ bb,
                          float* __restrict__ y) {
  const long row = blockIdx.x;
  const int lane = threadIdx.x;
  const float* xr = x + row * DD;
  const float* rr = r ? r + row * DD : nullptr;
  float v[8];
  float s = 0.0f, s2 = 0.0f;
#pragma unroll
  for (int j = 0; j < 8; ++j) {
    const int c = lane + j * 32;
    float t = xr[c] + (rr ? rr[c] : 0.0f);
    v[j] = t; s += t; s2 += t * t;
  }
  for (int m = 16; m >= 1; m >>= 1) { s += __shfl_xor(s, m, 32); s2 += __shfl_xor(s2, m, 32); }
  const float mean = s * (1.0f / DD);
  const float var  = s2 * (1.0f / DD) - mean * mean;
  const float inv  = rsqrtf(var + 1e-5f);
#pragma unroll
  for (int j = 0; j < 8; ++j) {
    const int c = lane + j * 32;
    y[row * DD + c] = (v[j] - mean) * inv * w[c] + bb[c];
  }
}

// ---------------- elementwise add ----------------
__global__ void ew_add_kernel(const float* __restrict__ a, const float* __restrict__ b,
                              float* __restrict__ c, long n) {
  const long i = (long)blockIdx.x * blockDim.x + threadIdx.x;
  if (i < n) c[i] = a[i] + b[i];
}

// ---------------- deformable sampling: wave per (b,q,h), lane = channel ----------------
__global__ void deform_kernel(const float* __restrict__ refp, const float* __restrict__ off,
                              const float* __restrict__ awr, const float* __restrict__ valp,
                              float* __restrict__ dout) {
  const int q = blockIdx.x;
  const int b = blockIdx.y;
  const int h = threadIdx.y;
  const int lane = threadIdx.x;   // d in 0..31
  const int hls[LL]    = {128, 64, 32, 16};
  const int wls[LL]    = {128, 64, 32, 16};
  const int starts[LL] = {0, 16384, 20480, 21504};

  // softmax over the 16 (l,p) logits, computed on lanes 0..15
  const float* awrow = awr + ((long)b * NQ + q) * (HH * LL * PP) + h * (LL * PP);
  float logit = (lane < 16) ? awrow[lane] : -1e30f;
  float mx = logit;
  for (int m = 8; m >= 1; m >>= 1) mx = fmaxf(mx, __shfl_xor(mx, m, 32));
  float ex = (lane < 16) ? expf(logit - mx) : 0.0f;
  float sm = ex;
  for (int m = 8; m >= 1; m >>= 1) sm += __shfl_xor(sm, m, 32);
  const float wnorm = ex / fmaxf(sm, 1e-30f);

  const float* offrow = off + ((long)b * NQ + q) * (HH * LL * PP * 2) + h * (LL * PP * 2);
  const float* vbase  = valp + (long)b * LV * DD + h * DH + lane;
  float acc = 0.0f;
#pragma unroll
  for (int l = 0; l < LL; ++l) {
    const int hl = hls[l], wl = wls[l], st = starts[l];
    const float refx = refp[(((long)b * NQ + q) * LL + l) * 2 + 0];
    const float refy = refp[(((long)b * NQ + q) * LL + l) * 2 + 1];
#pragma unroll
    for (int p = 0; p < PP; ++p) {
      const float ox = offrow[(l * PP + p) * 2 + 0];
      const float oy = offrow[(l * PP + p) * 2 + 1];
      const float x = (refx + ox / (float)wl) * (float)wl - 0.5f;
      const float y = (refy + oy / (float)hl) * (float)hl - 0.5f;
      const float x0f = floorf(x), y0f = floorf(y);
      const float lx = x - x0f, ly = y - y0f;
      const int x0 = (int)x0f, y0 = (int)y0f;
      float sval = 0.0f;
#pragma unroll
      for (int cy = 0; cy < 2; ++cy) {
#pragma unroll
        for (int cx = 0; cx < 2; ++cx) {
          const int xi = x0 + cx, yi = y0 + cy;
          const bool valid = (xi >= 0) && (xi < wl) && (yi >= 0) && (yi < hl);
          const int xc = min(max(xi, 0), wl - 1);
          const int yc = min(max(yi, 0), hl - 1);
          const float vv = vbase[(long)(st + yc * wl + xc) * DD];
          const float wgt = (cx ? lx : (1.0f - lx)) * (cy ? ly : (1.0f - ly));
          sval += valid ? vv * wgt : 0.0f;
        }
      }
      acc += __shfl(wnorm, l * PP + p, 32) * sval;
    }
  }
  dout[((long)b * NQ + q) * DD + h * DH + lane] = acc;
}

// ---------------- host side ----------------
extern "C" void kernel_launch(void* const* d_in, const int* in_sizes, int n_in,
                              void* d_out, int out_size, void* d_ws, size_t ws_size,
                              hipStream_t stream) {
  const float* query   = (const float*)d_in[0];
  const float* qpos    = (const float*)d_in[1];
  const float* refp    = (const float*)d_in[2];
  const float* value   = (const float*)d_in[3];
  const float* rel     = (const float*)d_in[4];
  const float* sa_qw   = (const float*)d_in[5];
  const float* sa_qb   = (const float*)d_in[6];
  const float* sa_kw   = (const float*)d_in[7];
  const float* sa_kb   = (const float*)d_in[8];
  const float* sa_vw   = (const float*)d_in[9];
  const float* sa_vb   = (const float*)d_in[10];
  const float* sa_ow   = (const float*)d_in[11];
  const float* sa_ob   = (const float*)d_in[12];
  const float* sa_nw   = (const float*)d_in[13];
  const float* sa_nb   = (const float*)d_in[14];
  const float* norm2_w = (const float*)d_in[15];
  const float* norm2_b = (const float*)d_in[16];
  const float* off_w   = (const float*)d_in[17];
  const float* off_b   = (const float*)d_in[18];
  const float* aw_w    = (const float*)d_in[19];
  const float* aw_b    = (const float*)d_in[20];
  const float* vp_w    = (const float*)d_in[21];
  const float* vp_b    = (const float*)d_in[22];
  const float* op_w    = (const float*)d_in[23];
  const float* op_b    = (const float*)d_in[24];
  const float* norm1_w = (const float*)d_in[25];
  const float* norm1_b = (const float*)d_in[26];
  const float* w1      = (const float*)d_in[27];
  const float* b1      = (const float*)d_in[28];
  const float* w2      = (const float*)d_in[29];
  const float* b2      = (const float*)d_in[30];
  const float* norm3_w = (const float*)d_in[31];
  const float* norm3_b = (const float*)d_in[32];
  float* out = (float*)d_out;

  // scratch layout (floats)
  const size_t NQD  = (size_t)BNQ * DD;           // 1,843,200
  const size_t SZ_S = (size_t)BB * HH * NQ * NQP; // 53,452,800
  float* ws   = (float*)d_ws;
  float* QWP  = ws + 0 * NQD;     // later reused as X1P
  float* Qb   = ws + 1 * NQD;     // later reused as T1 (o-proj out)
  float* Kb   = ws + 2 * NQD;     // later reused as Q2
  float* Vb   = ws + 3 * NQD;     // later reused as OFF
  float* Ob   = ws + 4 * NQD;     // later reused as D2
  float* X1   = ws + 5 * NQD;     // later reused as F2
  float* X2   = ws + 6 * NQD;
  float* H1   = ws + 7 * NQD;     // 7200*1024 = 4*NQD
  float* Sb   = ws + 11 * NQD;    // size SZ_S
  float* AW   = Sb;                       // reused after attention is done
  float* VALP = Sb + 2000000;             // 44,564,480 floats
  float* DOUT = Sb + 47000000;            // 1,843,200 floats
  float* X1P  = QWP;
  float* T1   = Qb;
  float* Q2   = Kb;
  float* OFFp = Vb;
  float* D2   = Ob;
  float* F2   = X1;

  // bf16 region after the score buffer (16B-aligned offsets)
  __bf16* bf = (__bf16*)(ws + 11 * NQD + SZ_S);
  __bf16* wt_qw  = bf + 0;        // 256*256
  __bf16* wt_kw  = bf + 65536;
  __bf16* wt_vw  = bf + 131072;
  __bf16* wt_ow  = bf + 196608;
  __bf16* wt_off = bf + 262144;   // 256*256
  __bf16* wt_aw  = bf + 327680;   // 256*128
  __bf16* wt_vp  = bf + 360448;   // 256*256
  __bf16* wt_op  = bf + 425984;   // 256*256
  __bf16* wt_w1  = bf + 491520;   // 256*1024
  __bf16* wt_w2  = bf + 753664;   // 1024*256
  __bf16* VT     = bf + 1015808;  // 64*32*928 bf16
  (void)ws_size; (void)in_sizes; (void)n_in; (void)out_size;

  auto tw = [&](const float* W, __bf16* WT, int K, int N) {
    transpose_w_kernel<<<dim3((K + 31) / 32, (N + 7) / 8), dim3(32, 8), 0, stream>>>(W, WT, K, N);
  };
  auto gemm256 = [&](const float* A, const __bf16* WT, const float* bias, float* C,
                     int M, int N, int relu) {
    gemm_bias_kernel<256><<<dim3(M / 16, N / 64), 32, 0, stream>>>(A, WT, bias, C, M, N, relu);
  };

  // one-time weight transposes (bf16)
  tw(sa_qw, wt_qw, DD, DD);   tw(sa_kw, wt_kw, DD, DD);
  tw(sa_vw, wt_vw, DD, DD);   tw(sa_ow, wt_ow, DD, DD);
  tw(off_w, wt_off, DD, DD);  tw(aw_w,  wt_aw, DD, HH * LL * PP);
  tw(vp_w,  wt_vp, DD, DD);   tw(op_w,  wt_op, DD, DD);
  tw(w1,    wt_w1, DD, DFF);  tw(w2,    wt_w2, DFF, DD);

  // --- relation self-attention ---
  ew_add_kernel<<<(unsigned)((NQD + 255) / 256), 256, 0, stream>>>(query, qpos, QWP, (long)NQD);
  gemm256(QWP,   wt_qw, sa_qb, Qb, BNQ, DD, 0);
  gemm256(QWP,   wt_kw, sa_kb, Kb, BNQ, DD, 0);
  gemm256(query, wt_vw, sa_vb, Vb, BNQ, DD, 0);
  pack_v_kernel<<<dim3(NQP / 32, BB * HH), dim3(32, 8), 0, stream>>>(Vb, VT);
  attn_scores_kernel<<<dim3((NQ + 15) / 16, NQP / 32, BB * HH), 32, 0, stream>>>(Qb, Kb, Sb);
  attn_softmax_kernel<<<dim3(NQ, BB * HH), 256, 0, stream>>>(Sb, rel);
  attn_v_kernel<<<dim3((NQ + 15) / 16, 1, BB * HH), 32, 0, stream>>>(Sb, VT, Ob);
  gemm256(Ob, wt_ow, sa_ob, T1, BNQ, DD, 0);
  ln_kernel<<<BNQ, 32, 0, stream>>>(T1, nullptr, sa_nw, sa_nb, Q2);
  ln_kernel<<<BNQ, 32, 0, stream>>>(query, Q2, norm2_w, norm2_b, X1);

  // --- deformable attention ---
  ew_add_kernel<<<(unsigned)((NQD + 255) / 256), 256, 0, stream>>>(X1, qpos, X1P, (long)NQD);
  gemm256(X1P,   wt_off, off_b, OFFp, BNQ, HH * LL * PP * 2, 0);
  gemm256(X1P,   wt_aw,  aw_b,  AW,   BNQ, HH * LL * PP,     0);
  gemm256(value, wt_vp,  vp_b,  VALP, BLV, DD,               0);
  deform_kernel<<<dim3(NQ, BB), dim3(32, HH), 0, stream>>>(refp, OFFp, AW, VALP, DOUT);
  gemm256(DOUT, wt_op, op_b, D2, BNQ, DD, 0);
  ln_kernel<<<BNQ, 32, 0, stream>>>(X1, D2, norm1_w, norm1_b, X2);

  // --- FFN ---
  gemm256(X2, wt_w1, b1, H1, BNQ, DFF, 1);
  gemm_bias_kernel<1024><<<dim3(BNQ / 16, DD / 64), 32, 0, stream>>>(H1, wt_w2, b2, F2, BNQ, DD, 0);
  ln_kernel<<<BNQ, 32, 0, stream>>>(X2, F2, norm3_w, norm3_b, out);
}
